// MoELayerStacks_31482110280255
// MI455X (gfx1250) — compile-verified, matrix operand order checked by reading, not run on hardware
//
#include <hip/hip_runtime.h>
#include <hip/hip_bf16.h>
#include <stdint.h>

#define E_  8
#define L2_ 15
#define L3_ 32
#define D_  3072
#define R_  256
#define B_  16384

typedef __bf16 bf16_t;
typedef __attribute__((ext_vector_type(16))) __bf16 v16bf;
typedef __attribute__((ext_vector_type(8)))  float  v8f;
typedef int gvec4i __attribute__((vector_size(16)));   // matches builtin's V4i

#define KSTEPS   (D_ / 32)          // 96 k-blocks of 32
#define NTILES   8                  // 128 output cols / 16
#define BLOCK_WAVES 2               // 2 wave32 per block
#define ROWS_PER_WAVE 32            // 2 M-tiles per wave
#define ROWS_PER_BLOCK (BLOCK_WAVES * ROWS_PER_WAVE)   // 64
#define KSLICE_BYTES 8192           // one k-step of packed B: 8 frags * 32 lanes * 32B

// ---- workspace layout (bytes) ----
#define WS_ACCUM   0                       // 19 floats (pad to 128B)
#define WS_EIDX    128                     // B_ ints   (64KB)
#define WS_MBIAS   (128 + B_ * 4)          // 128 floats
#define WS_BPACK   66304                   // 128*3072 bf16 = 786432 B, 256-aligned

#define AS1 __attribute__((address_space(1)))
#define AS3 __attribute__((address_space(3)))

#if defined(__HIP_DEVICE_COMPILE__) && __has_builtin(__builtin_amdgcn_global_load_async_to_lds_b128)
#define HAVE_ASYNC 1
#else
#define HAVE_ASYNC 0
#endif

#if HAVE_ASYNC
#if __has_builtin(__builtin_amdgcn_s_wait_asynccnt)
#define ASYNC_FENCE() __builtin_amdgcn_s_wait_asynccnt(0)
#else
#define ASYNC_FENCE() asm volatile("s_wait_asynccnt 0" ::: "memory")
#endif
#else
#define ASYNC_FENCE() ((void)0)
#endif

__device__ __forceinline__ float clamp01(float v) {
    return fminf(fmaxf(v, 0.0f), 1.0f);
}

// Copy 128 contiguous bytes global -> LDS (per thread) for B staging.
__device__ __forceinline__ void stage_b128x8(const char* __restrict__ src, char* __restrict__ dst) {
#if HAVE_ASYNC
#pragma unroll
    for (int j = 0; j < 8; ++j)
        __builtin_amdgcn_global_load_async_to_lds_b128(
            (AS1 gvec4i*)(uintptr_t)(src + j * 16),
            (AS3 gvec4i*)(uintptr_t)(dst + j * 16), 0, 0);
#else
    const uint4* s4 = (const uint4*)src;
    uint4* d4 = (uint4*)dst;
#pragma unroll
    for (int j = 0; j < 8; ++j) d4[j] = s4[j];
#endif
}

__device__ __forceinline__ v16bf load_a_frag(const float* __restrict__ p) {
    const float4* ap = reinterpret_cast<const float4*>(p);
    float4 q0 = ap[0], q1 = ap[1], q2 = ap[2], q3 = ap[3];
    v16bf a;
    a[0]=(bf16_t)q0.x;  a[1]=(bf16_t)q0.y;  a[2]=(bf16_t)q0.z;  a[3]=(bf16_t)q0.w;
    a[4]=(bf16_t)q1.x;  a[5]=(bf16_t)q1.y;  a[6]=(bf16_t)q1.z;  a[7]=(bf16_t)q1.w;
    a[8]=(bf16_t)q2.x;  a[9]=(bf16_t)q2.y;  a[10]=(bf16_t)q2.z; a[11]=(bf16_t)q2.w;
    a[12]=(bf16_t)q3.x; a[13]=(bf16_t)q3.y; a[14]=(bf16_t)q3.z; a[15]=(bf16_t)q3.w;
    return a;
}

// ---------------- kernel 0: zero stats accumulators ----------------
__global__ void zero_accum_kernel(float* accum) {
    if (threadIdx.x < 32) accum[threadIdx.x] = 0.0f;
}

// ---------------- kernel 1: merge + pack weights into WMMA-B fragment layout ----
// frag(kb,nt): lane = (khalf<<4)|n holds col nt*16+(lane&15), K = kb*32+(lane>>4)*16+j.
__global__ void pack_weights_kernel(const float* __restrict__ l1_w,
                                    const float* __restrict__ l1f_w,
                                    const float* __restrict__ l1_b,
                                    const float* __restrict__ l1f_b,
                                    bf16_t* __restrict__ bpack,
                                    float*  __restrict__ mbias) {
    int idx = blockIdx.x * blockDim.x + threadIdx.x;      // 0 .. 96*8*32-1
    if (idx >= KSTEPS * NTILES * 32) return;
    int lane = idx & 31;
    int nt   = (idx >> 5) & 7;
    int kb   = idx >> 8;
    int o    = nt * 16 + (lane & 15);                     // 0..127 = e*16 + oo
    int kbase = kb * 32 + (lane >> 4) * 16;
    const float* wrow = l1_w  + (size_t)o * D_;
    const float* frow = l1f_w + (size_t)(o & 15) * D_;
    bf16_t* dst = bpack + (size_t)idx * 16;
#pragma unroll
    for (int j = 0; j < 16; ++j)
        dst[j] = (bf16_t)(wrow[kbase + j] + frow[kbase + j]);
    if (idx < 128) mbias[idx] = l1_b[idx] + l1f_b[idx & 15];
}

// ---------------- kernel 2: router (logits, softmax stats, argmax) ----------------
__global__ __launch_bounds__(256) void router_kernel(const float* __restrict__ rin,
                                                     const float* __restrict__ rw,
                                                     const float* __restrict__ rb,
                                                     int*   __restrict__ eidx,
                                                     float* __restrict__ accum) {
    __shared__ float s_rw[E_ * R_];    // 8KB
    __shared__ float s_stats[19];
    int tid = threadIdx.x;
    for (int i = tid; i < E_ * R_; i += 256) s_rw[i] = rw[i];
    if (tid < 19) s_stats[tid] = 0.0f;
    __syncthreads();

    int wave = tid >> 5, lane = tid & 31;
    int b = blockIdx.x * 8 + wave;
    const float* x = rin + (size_t)b * R_;

    float acc[E_];
#pragma unroll
    for (int e = 0; e < E_; ++e) acc[e] = 0.0f;
    for (int i = lane; i < R_; i += 32) {
        float xv = x[i];
#pragma unroll
        for (int e = 0; e < E_; ++e) acc[e] = fmaf(xv, s_rw[e * R_ + i], acc[e]);
    }
#pragma unroll
    for (int off = 16; off > 0; off >>= 1)
#pragma unroll
        for (int e = 0; e < E_; ++e) acc[e] += __shfl_xor(acc[e], off, 32);

    if (lane == 0) {
        float logits[E_], mx = -3.4e38f;
        int am = 0;
#pragma unroll
        for (int e = 0; e < E_; ++e) {
            logits[e] = acc[e] + rb[e];
            if (logits[e] > mx) { mx = logits[e]; am = e; }
        }
        float se = 0.0f, p[E_];
#pragma unroll
        for (int e = 0; e < E_; ++e) { p[e] = __expf(logits[e] - mx); se += p[e]; }
        float inv = 1.0f / se;
        float lse = mx + __logf(se);
        float ent = 0.0f;
#pragma unroll
        for (int e = 0; e < E_; ++e) {
            p[e] *= inv;
            ent -= p[e] * __logf(fmaxf(p[e], 1e-9f));
        }
        eidx[b] = am;
        atomicAdd(&s_stats[am], 1.0f);
#pragma unroll
        for (int e = 0; e < E_; ++e) atomicAdd(&s_stats[8 + e], p[e]);
        atomicAdd(&s_stats[16], lse * lse);
        atomicAdd(&s_stats[17], ent);
        atomicAdd(&s_stats[18], p[am]);
    }
    __syncthreads();
    if (tid < 19) atomicAdd(&accum[tid], s_stats[tid]);
}

// ---------------- kernel 3: main GEMM (bf16 WMMA, LDS-staged B) + one-hot tail ----
// 64 threads = 2 wave32, each wave: 32 rows x 128 cols (16 C tiles).
// B k-slice (8KB) double-buffered in LDS via async global->LDS copies.
__global__ __launch_bounds__(64) void moe_gemm_kernel(const float*  __restrict__ X,
                                                      const bf16_t* __restrict__ bpack,
                                                      const float*  __restrict__ mbias,
                                                      const float*  __restrict__ l2w,
                                                      const float*  __restrict__ l2b,
                                                      const float*  __restrict__ ow,
                                                      const float*  __restrict__ ob,
                                                      const int*    __restrict__ eidx,
                                                      float* __restrict__ out) {
    extern __shared__ char smem[];
    char*  bbuf  = smem;                               // 2 * 8KB B slices
    float* s_l1c = (float*)(smem + 2 * KSLICE_BYTES);  // 64*128 floats (32KB)
    float* s_l2w = s_l1c + ROWS_PER_BLOCK * 128;       // 256*30 = 7680
    float* s_l2b = s_l2w + E_ * L3_ * 2 * L2_;         // 256
    float* s_ow  = s_l2b + E_ * L3_;                   // 256
    float* s_mb  = s_ow + E_ * L3_;                    // 128
    float* s_ob  = s_mb + 128;                         // 8

    int tid = threadIdx.x;
    for (int i = tid; i < E_ * L3_ * 2 * L2_; i += 64) s_l2w[i] = l2w[i];
    for (int i = tid; i < E_ * L3_; i += 64) { s_l2b[i] = l2b[i]; s_ow[i] = ow[i]; }
    for (int i = tid; i < 128; i += 64) s_mb[i] = mbias[i];
    if (tid < 8) s_ob[tid] = ob[tid];

    const char* bsrc = (const char*)bpack;
    // prologue: stage k-slice 0 into buffer 0
    stage_b128x8(bsrc + (size_t)tid * 128, bbuf + tid * 128);
    ASYNC_FENCE();
    __syncthreads();

    int wave = tid >> 5, lane = tid & 31;
    int m = lane & 15, kh = lane >> 4;                 // row-in-tile, K-half
    size_t rowbase = (size_t)blockIdx.x * ROWS_PER_BLOCK + wave * ROWS_PER_WAVE;
    const float* arow0 = X + (rowbase + m) * D_ + kh * 16;
    const float* arow1 = arow0 + (size_t)16 * D_;

    v8f c[16];
#pragma unroll
    for (int t = 0; t < 16; ++t) c[t] = (v8f){0,0,0,0,0,0,0,0};

#pragma unroll 1
    for (int kb = 0; kb < KSTEPS; ++kb) {
        int cur = kb & 1;
        if (kb + 1 < KSTEPS) {
            // overlap: DMA next B k-slice into the other buffer
            stage_b128x8(bsrc + (size_t)(kb + 1) * KSLICE_BYTES + tid * 128,
                         bbuf + (cur ^ 1) * KSLICE_BYTES + tid * 128);
            __builtin_prefetch(arow0 + (size_t)(kb + 1) * 32, 0, 0);
            __builtin_prefetch(arow1 + (size_t)(kb + 1) * 32, 0, 0);
        }
        v16bf a0 = load_a_frag(arow0 + (size_t)kb * 32);
        v16bf a1 = load_a_frag(arow1 + (size_t)kb * 32);
        const char* bc = bbuf + cur * KSLICE_BYTES + lane * 32;
#pragma unroll
        for (int nt = 0; nt < NTILES; ++nt) {
            v16bf bf = *(const v16bf*)(bc + nt * 1024);
            c[nt] = __builtin_amdgcn_wmma_f32_16x16x32_bf16(
                        false, a0, false, bf, (short)0, c[nt], false, false);
            c[8 + nt] = __builtin_amdgcn_wmma_f32_16x16x32_bf16(
                        false, a1, false, bf, (short)0, c[8 + nt], false, false);
        }
        ASYNC_FENCE();           // next slice landed
        __syncthreads();         // everyone done reading cur before it is reused
    }

    // Spill C (32 rows x 128 cols per wave) into LDS.
    // C layout: lane l holds col (l&15) of tile, rows v + 8*(l>>4), v=0..7.
    float* my = s_l1c + wave * ROWS_PER_WAVE * 128;
#pragma unroll
    for (int rt = 0; rt < 2; ++rt)
#pragma unroll
        for (int nt = 0; nt < NTILES; ++nt) {
            int col = nt * 16 + m;
#pragma unroll
            for (int v = 0; v < 8; ++v)
                my[(rt * 16 + v + 8 * kh) * 128 + col] = c[rt * 8 + nt][v];
        }
    __syncthreads();

    // One-hot epilogue: forward gate == hard one-hot; 1 lane per row.
    int lr = lane;                                     // 0..31
    size_t gb = rowbase + lr;
    int ei = eidx[gb];
    const float* lrow = my + lr * 128 + ei * 16;

    float x[16];
#pragma unroll
    for (int oo = 0; oo < 16; ++oo) x[oo] = lrow[oo] + s_mb[ei * 16 + oo];
    float xo = x[L2_];

    float l1x[2 * L2_];
#pragma unroll
    for (int i = 0; i < L2_; ++i) {
        l1x[i]       = clamp01(x[i] * x[i] * (255.0f / 256.0f));
        l1x[L2_ + i] = clamp01(x[i]);
    }

    float res = 0.0f;
#pragma unroll
    for (int o = 0; o < L3_; ++o) {
        const float* wrow = s_l2w + (ei * L3_ + o) * (2 * L2_);
        float acc = s_l2b[ei * L3_ + o];
#pragma unroll
        for (int i = 0; i < 2 * L2_; ++i) acc = fmaf(l1x[i], wrow[i], acc);
        res = fmaf(clamp01(acc), s_ow[ei * L3_ + o], res);
    }
    out[gb] = res + s_ob[ei] + xo;
}

// ---------------- kernel 4: finalize scalar losses ----------------
__global__ void finalize_kernel(const float* __restrict__ accum, float* __restrict__ tail) {
    if (threadIdx.x != 0 || blockIdx.x != 0) return;
    const float invB = 1.0f / (float)B_;
    float frac[E_], avg[E_], aux = 0.0f;
#pragma unroll
    for (int e = 0; e < E_; ++e) {
        frac[e] = accum[e] * invB;
        avg[e]  = accum[8 + e] * invB;
        aux += frac[e] * avg[e];
    }
    aux *= (float)E_;
    float z    = accum[16] * invB;
    float nent = accum[17] * invB / __logf((float)E_);
    float top1 = accum[18] * invB;
    tail[0] = 0.01f * aux + 0.001f * z;   // router_loss
    tail[1] = aux;
    tail[2] = z;
#pragma unroll
    for (int e = 0; e < E_; ++e) { tail[3 + e] = frac[e]; tail[11 + e] = avg[e]; }
    tail[19] = nent;
    tail[20] = top1;
}

extern "C" void kernel_launch(void* const* d_in, const int* in_sizes, int n_in,
                              void* d_out, int out_size, void* d_ws, size_t ws_size,
                              hipStream_t stream) {
    const float* expert_input = (const float*)d_in[0];
    const float* router_input = (const float*)d_in[1];
    const float* router_w     = (const float*)d_in[2];
    const float* router_b     = (const float*)d_in[3];
    const float* l1_w         = (const float*)d_in[4];
    const float* l1_b         = (const float*)d_in[5];
    const float* l1f_w        = (const float*)d_in[6];
    const float* l1f_b        = (const float*)d_in[7];
    const float* l2_w         = (const float*)d_in[8];
    const float* l2_b         = (const float*)d_in[9];
    const float* out_w        = (const float*)d_in[10];
    const float* out_b        = (const float*)d_in[11];

    float*  out   = (float*)d_out;                          // [B] then 21 scalars
    char*   ws    = (char*)d_ws;
    float*  accum = (float*)(ws + WS_ACCUM);
    int*    eidx  = (int*)(ws + WS_EIDX);
    float*  mbias = (float*)(ws + WS_MBIAS);
    bf16_t* bpack = (bf16_t*)(ws + WS_BPACK);

    zero_accum_kernel<<<1, 32, 0, stream>>>(accum);

    pack_weights_kernel<<<(KSTEPS * NTILES * 32 + 255) / 256, 256, 0, stream>>>(
        l1_w, l1f_w, l1_b, l1f_b, bpack, mbias);

    router_kernel<<<B_ / 8, 256, 0, stream>>>(
        router_input, router_w, router_b, eidx, accum);

    size_t smem = 2 * KSLICE_BYTES +
                  (size_t)(ROWS_PER_BLOCK * 128 + E_ * L3_ * 2 * L2_ +
                           2 * E_ * L3_ + 128 + 8) * sizeof(float);   // ~82.5 KB
    moe_gemm_kernel<<<B_ / ROWS_PER_BLOCK, 64, smem, stream>>>(
        expert_input, bpack, mbias, l2_w, l2_b, out_w, out_b, eidx, out);

    finalize_kernel<<<1, 32, 0, stream>>>(accum, out + B_);
}
